// LlamaAttention_55576876810531
// MI455X (gfx1250) — compile-verified, matrix operand order checked by reading, not run on hardware
//
#include <hip/hip_runtime.h>
#include <stdint.h>

// ---------------------------------------------------------------------------
// bf16 WMMA GEMM pipeline for the factorized-Llama block on gfx1250 (wave32).
// C = A @ W^T (torch Linear convention), bf16 inputs, f32 WMMA accumulation.
// Fully in-bounds blocks stage tiles with GLOBAL_LOAD_ASYNC_TO_LDS_B128
// (ASYNCcnt); edge blocks use clamped-address + select (no EXEC juggling).
// The async/masked choice is hoisted out of the K loop (block-uniform).
// ---------------------------------------------------------------------------

#define BM 128
#define BN 128
#define BK 32
#define LDST 40   // LDS row stride in halves: 80 bytes, 16B-aligned fragments

typedef __bf16 v16bf __attribute__((ext_vector_type(16)));
typedef __bf16 v8bf  __attribute__((ext_vector_type(8)));
typedef float  v8f   __attribute__((ext_vector_type(8)));
typedef int    v4i   __attribute__((ext_vector_type(4)));

#if defined(__AMDGCN__) && __has_builtin(__builtin_amdgcn_global_load_async_to_lds_b128)
#define HAVE_ASYNC_LDS 1
#else
#define HAVE_ASYNC_LDS 0
#endif

#if HAVE_ASYNC_LDS
__device__ __forceinline__ void async_copy16(const unsigned short* g, unsigned short* l) {
  __builtin_amdgcn_global_load_async_to_lds_b128(
      (__attribute__((address_space(1))) v4i*)(v4i*)(void*)const_cast<unsigned short*>(g),
      (__attribute__((address_space(3))) v4i*)(v4i*)(void*)l, 0, 0);
}
__device__ __forceinline__ void wait_async0() {
#if __has_builtin(__builtin_amdgcn_s_wait_asynccnt)
  __builtin_amdgcn_s_wait_asynccnt(0);
#else
  asm volatile("s_wait_asynccnt 0x0" ::: "memory");
#endif
}
#endif

__device__ __forceinline__ unsigned short f2bf(float f) {
  unsigned u = __float_as_uint(f);
  u += 0x7FFFu + ((u >> 16) & 1u);   // round-to-nearest-even
  return (unsigned short)(u >> 16);
}

// Branchless 16-byte row-chunk stage: clamp row, load, mask to zero if OOB row.
__device__ __forceinline__ void stage16_masked(const unsigned short* base, long rowStride,
                                               int row, int nrows, long colOff,
                                               unsigned short* dst) {
  int rc = row < nrows ? row : (nrows - 1);
  unsigned msk = row < nrows ? ~0u : 0u;
  const unsigned* src = (const unsigned*)(const void*)(base + (long)rc * rowStride + colOff);
  unsigned* d = (unsigned*)(void*)dst;
#pragma unroll
  for (int e = 0; e < 4; ++e) d[e] = src[e] & msk;
}

// Branchless per-element K-tail stage (clamped address + select).
__device__ __forceinline__ void stage8_tail(const unsigned short* base, long rowStride,
                                            int row, int nrows, int k0, int ck, int K,
                                            unsigned short* dst) {
  int rc = row < nrows ? row : (nrows - 1);
  const unsigned short* rp = base + (long)rc * rowStride;
#pragma unroll
  for (int e = 0; e < 8; ++e) {
    int gk = k0 + ck + e;
    int gkc = gk < K ? gk : (K - 1);
    unsigned short v = rp[gkc];
    dst[e] = (row < nrows && gk < K) ? v : (unsigned short)0;
  }
}

// Load one lane's 16x32 bf16 fragment from LDS per the CDNA5 16-bit layout:
// lanes 0-15: K={0..7}u{16..23}; lanes 16-31: K={8..15}u{24..31}.
// p points at &tile[row][(lane>>4)*8].
__device__ __forceinline__ v16bf ldfrag(const unsigned short* p) {
  v8bf lo = *(const v8bf*)(const void*)p;
  v8bf hi = *(const v8bf*)(const void*)(p + 16);
  v16bf r;
#pragma unroll
  for (int e = 0; e < 8; ++e) { r[e] = lo[e]; r[e + 8] = hi[e]; }
  return r;
}

// One BK-step of WMMA: 2 (M) x 4 (N) 16x16 tiles per wave.
__device__ __forceinline__ void wmma_step(const unsigned short (*As)[LDST],
                                          const unsigned short (*Bs)[LDST],
                                          v8f (&acc)[2][4],
                                          int wm, int wn, int half, int idx) {
  v16bf af[2], bfr[4];
#pragma unroll
  for (int mi = 0; mi < 2; ++mi)
    af[mi] = ldfrag(&As[wm + mi * 16 + idx][half * 8]);
#pragma unroll
  for (int ni = 0; ni < 4; ++ni)
    bfr[ni] = ldfrag(&Bs[wn + ni * 16 + idx][half * 8]);
#pragma unroll
  for (int mi = 0; mi < 2; ++mi)
#pragma unroll
    for (int ni = 0; ni < 4; ++ni)
      acc[mi][ni] = __builtin_amdgcn_wmma_f32_16x16x32_bf16(
          false, af[mi], false, bfr[ni], (short)0, acc[mi][ni], false, false);
}

// OUT_BF16: write C as bf16 (feeds next GEMM) or f32 (logits / final output).
// NT: W is [N,K] row-major (Linear weight).  !NT: B is [K,N] row-major.
template <bool OUT_BF16, bool NT>
__global__ __launch_bounds__(256) void gemm_bf16_wmma(
    const unsigned short* __restrict__ A,   // [M,K] bf16 row-major
    const unsigned short* __restrict__ Bm,  // NT: [N,K]; NN: [K,N]
    void* __restrict__ Cp,                  // [M,N] f32 or bf16
    int M, int N, int K,
    long sA, long sB, long sC)              // batch strides in elements
{
  __shared__ __align__(16) unsigned short As[BM][LDST];
  __shared__ __align__(16) unsigned short Bs[BN][LDST];

  const int tid  = threadIdx.x;
  const int lane = tid & 31;
  const int wave = tid >> 5;
  const int bm0  = blockIdx.y * BM;
  const int bn0  = blockIdx.x * BN;
  A  += (long)blockIdx.z * sA;
  Bm += (long)blockIdx.z * sB;
  const long cbase = (long)blockIdx.z * sC;

  const int wm   = (wave & 3) * 32;   // 4 wave-rows of 32
  const int wn   = (wave >> 2) * 64;  // 2 wave-cols of 64
  const int half = lane >> 4;
  const int idx  = lane & 15;

  // per-thread staging coordinates (shared by all variants)
  const int rA  = tid >> 2;           // 0..63 (+64 on second pass)
  const int ckA = (tid & 3) * 8;      // 16B chunk within BK
  const int kkB = tid >> 4;           // NN: 0..15 (+16 on second pass)
  const int nB  = (tid & 15) * 8;     // NN: 8-wide n chunk

  v8f zero = {};
  v8f acc[2][4];
#pragma unroll
  for (int i = 0; i < 2; ++i)
#pragma unroll
    for (int j = 0; j < 4; ++j) acc[i][j] = zero;

  const int Kfull = K & ~(BK - 1);
  const bool fullTiles = ((bm0 + BM) <= M) && (NT ? ((bn0 + BN) <= N) : ((bn0 + BN) <= N));

#if HAVE_ASYNC_LDS
  if (fullTiles) {
    // -------- async main loop: zero guards, pure CDNA5 async staging --------
    for (int k0 = 0; k0 < Kfull; k0 += BK) {
#pragma unroll
      for (int p = 0; p < 2; ++p) {
        int r = rA + p * 64;
        async_copy16(A + (long)(bm0 + r) * K + k0 + ckA, &As[r][ckA]);
      }
      if (NT) {
#pragma unroll
        for (int p = 0; p < 2; ++p) {
          int r = rA + p * 64;
          async_copy16(Bm + (long)(bn0 + r) * K + k0 + ckA, &Bs[r][ckA]);
        }
      } else {  // NN: transpose while staging (in-bounds by fullTiles)
#pragma unroll
        for (int p = 0; p < 2; ++p) {
          int kk = kkB + p * 16;
          const unsigned* src =
              (const unsigned*)(const void*)(Bm + (long)(k0 + kk) * N + bn0 + nB);
#pragma unroll
          for (int e = 0; e < 4; ++e) {
            unsigned w = src[e];
            Bs[nB + 2 * e][kk]     = (unsigned short)(w & 0xFFFFu);
            Bs[nB + 2 * e + 1][kk] = (unsigned short)(w >> 16);
          }
        }
      }
      wait_async0();
      __syncthreads();
      wmma_step(As, Bs, acc, wm, wn, half, idx);
      __syncthreads();
    }
  } else
#endif
  {
    // -------- masked main loop for edge blocks (branchless clamp+mask) --------
    for (int k0 = 0; k0 < Kfull; k0 += BK) {
#pragma unroll
      for (int p = 0; p < 2; ++p) {
        int r = rA + p * 64;
        stage16_masked(A, K, bm0 + r, M, k0 + ckA, &As[r][ckA]);
        if (NT) {
          stage16_masked(Bm, K, bn0 + r, N, k0 + ckA, &Bs[r][ckA]);
        } else {
          int kk = kkB + p * 16;
          int gn = bn0 + nB;
          int gnc = gn + 8 <= N ? gn : (N - 8);
          unsigned msk = (gn + 8 <= N) ? ~0u : 0u;
          const unsigned* src =
              (const unsigned*)(const void*)(Bm + (long)(k0 + kk) * N + gnc);
#pragma unroll
          for (int e = 0; e < 4; ++e) {
            unsigned w = src[e] & msk;
            Bs[nB + 2 * e][kk]     = (unsigned short)(w & 0xFFFFu);
            Bs[nB + 2 * e + 1][kk] = (unsigned short)(w >> 16);
          }
        }
      }
      __syncthreads();
      wmma_step(As, Bs, acc, wm, wn, half, idx);
      __syncthreads();
    }
  }

  // -------------------- K-tail epilogue (branchless selects) --------------------
  if (Kfull < K) {
#pragma unroll
    for (int p = 0; p < 2; ++p) {
      int r = rA + p * 64;
      stage8_tail(A, K, bm0 + r, M, Kfull, ckA, K, &As[r][ckA]);
      if (NT) {
        stage8_tail(Bm, K, bn0 + r, N, Kfull, ckA, K, &Bs[r][ckA]);
      } else {
        int kk = kkB + p * 16;
        int gk = Kfull + kk;
        int gkc = gk < K ? gk : (K - 1);
        const unsigned short* rp = Bm + (long)gkc * N;
#pragma unroll
        for (int e = 0; e < 8; ++e) {
          int gn = bn0 + nB + e;
          int gnec = gn < N ? gn : (N - 1);
          unsigned short v = rp[gnec];
          Bs[nB + e][kk] = (gk < K && gn < N) ? v : (unsigned short)0;
        }
      }
    }
    __syncthreads();
    wmma_step(As, Bs, acc, wm, wn, half, idx);
    __syncthreads();
  }

  // ---- store C (VGPR v: lanes 0-15 -> M=v, lanes 16-31 -> M=v+8; N=lane&15) ----
#pragma unroll
  for (int mi = 0; mi < 2; ++mi) {
#pragma unroll
    for (int ni = 0; ni < 4; ++ni) {
      int m0 = bm0 + wm + mi * 16 + half * 8;
      int n  = bn0 + wn + ni * 16 + idx;
      if (n < N) {
#pragma unroll
        for (int v = 0; v < 8; ++v) {
          int m = m0 + v;
          if (m < M) {
            if (OUT_BF16)
              ((unsigned short*)Cp)[cbase + (long)m * N + n] = f2bf(acc[mi][ni][v]);
            else
              ((float*)Cp)[cbase + (long)m * N + n] = acc[mi][ni][v];
          }
        }
      }
    }
  }
}

// ---------------------------------------------------------------------------

__global__ void cvt_f32_bf16(const float* __restrict__ in,
                             unsigned short* __restrict__ out, long n) {
  long i = (long)blockIdx.x * blockDim.x + threadIdx.x;
  if (i < n) out[i] = f2bf(in[i]);
}

// rope table: t[s, j<H/2] = sin(s * theta^(-2j/H)); t[s, j>=H/2] = cos(...)
__global__ void rope_bf16(unsigned short* __restrict__ t, int S, int Hd) {
  int s = blockIdx.x;
  int halfd = Hd >> 1;
  for (int j = threadIdx.x; j < Hd; j += blockDim.x) {
    int i = (j < halfd) ? j : (j - halfd);
    float freq = __powf(10000.0f, -(2.0f * (float)i) / (float)Hd);
    float ang  = (float)s * freq;
    float v    = (j < halfd) ? __sinf(ang) : __cosf(ang);
    t[(long)s * Hd + j] = f2bf(v);
  }
}

__global__ __launch_bounds__(256) void softmax_rows(
    const float* __restrict__ X, unsigned short* __restrict__ Ybf, int n) {
  __shared__ float red[256];
  const int tid = threadIdx.x;
  const float* x = X + (long)blockIdx.x * n;
  float m = -3.4e38f;
  for (int i = tid; i < n; i += 256) m = fmaxf(m, x[i]);
  red[tid] = m; __syncthreads();
  for (int s = 128; s > 0; s >>= 1) {
    if (tid < s) red[tid] = fmaxf(red[tid], red[tid + s]);
    __syncthreads();
  }
  m = red[0]; __syncthreads();
  float acc = 0.f;
  for (int i = tid; i < n; i += 256) acc += __expf(x[i] - m);
  red[tid] = acc; __syncthreads();
  for (int s = 128; s > 0; s >>= 1) {
    if (tid < s) red[tid] += red[tid + s];
    __syncthreads();
  }
  float inv = 1.0f / red[0];
  unsigned short* y = Ybf + (long)blockIdx.x * n;
  for (int i = tid; i < n; i += 256) y[i] = f2bf(__expf(x[i] - m) * inv);
}

// ---------------------------------------------------------------------------

extern "C" void kernel_launch(void* const* d_in, const int* in_sizes, int n_in,
                              void* d_out, int out_size, void* d_ws, size_t ws_size,
                              hipStream_t stream) {
  (void)in_sizes; (void)n_in; (void)out_size; (void)ws_size;
  const int Bz = 4, S = 4096, H = 768, R = 242, HK = 256;
  const long BS = (long)Bz * S;  // 16384 rows

  const float* x      = (const float*)d_in[0];
  // d_in[1..4] = q_low,q_high,k_low,k_high: dead code in the reference.
  const float* v_low  = (const float*)d_in[5];  // [R,H]
  const float* v_high = (const float*)d_in[6];  // [HK,R]
  const float* o_low  = (const float*)d_in[7];  // [R,HK]
  const float* o_high = (const float*)d_in[8];  // [H,R]

  char* ws = (char*)d_ws;
  size_t off = 0;
  auto take = [&](size_t bytes) -> char* {
    char* p = ws + off;
    off = (off + bytes + 255) & ~(size_t)255;
    return p;
  };
  unsigned short* xb    = (unsigned short*)take((size_t)BS * H * 2);
  unsigned short* vlw   = (unsigned short*)take((size_t)R * H * 2);
  unsigned short* vhw   = (unsigned short*)take((size_t)HK * R * 2);
  unsigned short* olw   = (unsigned short*)take((size_t)R * HK * 2);
  unsigned short* ohw   = (unsigned short*)take((size_t)H * R * 2);
  unsigned short* tmp1  = (unsigned short*)take((size_t)BS * R * 2);
  unsigned short* vmat  = (unsigned short*)take((size_t)BS * HK * 2);
  unsigned short* tpos  = (unsigned short*)take((size_t)S * H * 2);
  float*          logit = (float*)take((size_t)S * S * 4);
  unsigned short* attn  = (unsigned short*)take((size_t)S * S * 2);
  unsigned short* ctx   = (unsigned short*)take((size_t)BS * HK * 2);
  unsigned short* tmp2  = (unsigned short*)take((size_t)BS * R * 2);

  // bf16 casts of activations and live weights
  cvt_f32_bf16<<<(unsigned)((BS * H + 255) / 256), 256, 0, stream>>>(x, xb, BS * H);
  cvt_f32_bf16<<<(unsigned)(((long)R * H + 255) / 256), 256, 0, stream>>>(v_low, vlw, (long)R * H);
  cvt_f32_bf16<<<(unsigned)(((long)HK * R + 255) / 256), 256, 0, stream>>>(v_high, vhw, (long)HK * R);
  cvt_f32_bf16<<<(unsigned)(((long)R * HK + 255) / 256), 256, 0, stream>>>(o_low, olw, (long)R * HK);
  cvt_f32_bf16<<<(unsigned)(((long)H * R + 255) / 256), 256, 0, stream>>>(o_high, ohw, (long)H * R);
  rope_bf16<<<S, 256, 0, stream>>>(tpos, S, H);

  auto grid = [](long Mv, int Nv, int bz) {
    return dim3((unsigned)((Nv + BN - 1) / BN), (unsigned)((Mv + BM - 1) / BM), (unsigned)bz);
  };

  // v-projection chain: tmp1 = xb @ v_low^T ; vmat = tmp1 @ v_high^T
  gemm_bf16_wmma<true, true><<<grid(BS, R, 1), 256, 0, stream>>>(
      xb, vlw, tmp1, (int)BS, R, H, 0, 0, 0);
  gemm_bf16_wmma<true, true><<<grid(BS, HK, 1), 256, 0, stream>>>(
      tmp1, vhw, vmat, (int)BS, HK, R, 0, 0, 0);

  // attention scores: logits = t @ t^T  (input-independent, recomputed per call)
  gemm_bf16_wmma<false, true><<<grid(S, S, 1), 256, 0, stream>>>(
      tpos, tpos, logit, S, S, H, 0, 0, 0);
  softmax_rows<<<S, 256, 0, stream>>>(logit, attn, S);

  // ctx[b] = attn @ vmat[b]   (NN, batched; attn shared -> strideA = 0)
  gemm_bf16_wmma<true, false><<<grid(S, HK, Bz), 256, 0, stream>>>(
      attn, vmat, ctx, S, HK, S, 0L, (long)S * HK, (long)S * HK);

  // output chain: tmp2 = ctx @ o_low^T ; out = tmp2 @ o_high^T (f32)
  gemm_bf16_wmma<true, true><<<grid(BS, R, 1), 256, 0, stream>>>(
      ctx, olw, tmp2, (int)BS, R, HK, 0, 0, 0);
  gemm_bf16_wmma<false, true><<<grid(BS, H, 1), 256, 0, stream>>>(
      tmp2, ohw, d_out, (int)BS, H, R, 0, 0, 0);
}
